// gptJ_cloud_7799660609765
// MI455X (gfx1250) — compile-verified, matrix operand order checked by reading, not run on hardware
//
#include <hip/hip_runtime.h>
#include <hip/hip_bf16.h>
#include <cstdint>

#define BB 2
#define SS 2048
#define EE 4096
#define HH 16
#define DD 256
#define RR 64
#define MTOT (BB*SS)   // 4096 rows

typedef __attribute__((ext_vector_type(16))) __bf16 v16bf;
typedef __attribute__((ext_vector_type(8)))  float  v8f;
typedef __attribute__((ext_vector_type(4)))  int    v4i;

union Frag {
  v16bf v;
  uint4 u[2];
};

__device__ __forceinline__ uint16_t f2bf(float f) {
  union { float f; uint32_t u; } x; x.f = f;
  uint32_t r = x.u + 0x7FFFu + ((x.u >> 16) & 1u);   // round-to-nearest-even
  return (uint16_t)(r >> 16);
}

// --- async global->LDS staging (CDNA5 GLOBAL_LOAD_ASYNC_TO_LDS_B128) -------
#if defined(__has_builtin)
#if __has_builtin(__builtin_amdgcn_global_load_async_to_lds_b128) && \
    __has_builtin(__builtin_amdgcn_s_wait_asynccnt)
#define USE_ASYNC_LDS 1
#endif
#endif

typedef __attribute__((address_space(1))) v4i gv4i;   // global int4
typedef __attribute__((address_space(3))) v4i lv4i;   // LDS int4

__device__ __forceinline__ void copy16_g2l(uint16_t* ldsDst, const uint16_t* gSrc) {
#ifdef USE_ASYNC_LDS
  __builtin_amdgcn_global_load_async_to_lds_b128(
      (gv4i*)(uintptr_t)gSrc,
      (lv4i*)(uint32_t)(uintptr_t)ldsDst,
      0, 0);
#else
  *(uint4*)ldsDst = *(const uint4*)gSrc;
#endif
}

__device__ __forceinline__ void wait_async_le(int pending) {
#ifdef USE_ASYNC_LDS
  if (pending) __builtin_amdgcn_s_wait_asynccnt(6);
  else         __builtin_amdgcn_s_wait_asynccnt(0);
#else
  (void)pending;
#endif
}

// ---------------------------------------------------------------------------
// Kernel 1: LayerNorm, one wave32 per 4096-wide row, writes bf16 x_norm.
// ---------------------------------------------------------------------------
__global__ void __launch_bounds__(256)
ln_kernel(const float* __restrict__ x,
          const float* __restrict__ w,
          const float* __restrict__ b,
          uint16_t* __restrict__ xn)
{
  int wv = threadIdx.x >> 5;
  int lane = threadIdx.x & 31;
  int row = blockIdx.x * 8 + wv;
  const float4* xr = (const float4*)(x + (size_t)row * EE);
  float s = 0.f, s2 = 0.f;
#pragma unroll
  for (int i = 0; i < 32; ++i) {
    float4 v = xr[lane + i * 32];
    s  += v.x + v.y + v.z + v.w;
    s2 += v.x*v.x + v.y*v.y + v.z*v.z + v.w*v.w;
  }
#pragma unroll
  for (int m = 16; m >= 1; m >>= 1) {
    s  += __shfl_xor(s, m, 32);
    s2 += __shfl_xor(s2, m, 32);
  }
  float mean = s * (1.f / EE);
  float var  = s2 * (1.f / EE) - mean * mean;
  float rstd = rsqrtf(var + 1e-5f);
  const float4* wr = (const float4*)w;
  const float4* br = (const float4*)b;
  uint2* out = (uint2*)(xn + (size_t)row * EE);
#pragma unroll
  for (int i = 0; i < 32; ++i) {
    int idx = lane + i * 32;
    float4 v   = xr[idx];
    float4 wv4 = wr[idx];
    float4 bv4 = br[idx];
    float y0 = (v.x - mean) * rstd * wv4.x + bv4.x;
    float y1 = (v.y - mean) * rstd * wv4.y + bv4.y;
    float y2 = (v.z - mean) * rstd * wv4.z + bv4.z;
    float y3 = (v.w - mean) * rstd * wv4.w + bv4.w;
    uint2 p;
    p.x = (uint32_t)f2bf(y0) | ((uint32_t)f2bf(y1) << 16);
    p.y = (uint32_t)f2bf(y2) | ((uint32_t)f2bf(y3) << 16);
    out[idx] = p;
  }
}

// ---------------------------------------------------------------------------
// Kernel 2: f32 -> bf16 weight conversion (one float4 per thread).
// ---------------------------------------------------------------------------
__global__ void __launch_bounds__(256)
cvt_kernel(const float* __restrict__ src, uint16_t* __restrict__ dst)
{
  size_t i = (size_t)blockIdx.x * blockDim.x + threadIdx.x;
  float4 v = ((const float4*)src)[i];
  uint2 p;
  p.x = (uint32_t)f2bf(v.x) | ((uint32_t)f2bf(v.y) << 16);
  p.y = (uint32_t)f2bf(v.z) | ((uint32_t)f2bf(v.w) << 16);
  ((uint2*)dst)[i] = p;
}

// ---------------------------------------------------------------------------
// Kernel 3: projection GEMM  C[M,N] = A[M,K] * W[N,K]^T + bias
// bf16 WMMA, f32 accumulate. 128 threads = 4 waves, WG tile 128x64,
// wave tile 64x32. Double-buffered LDS staging fed by async global->LDS
// (ASYNCcnt-tracked) so each A/B slab is fetched from L2 once per WG and
// shared by 2 waves on each axis. LDS rows padded to 80B for bank spread.
// ---------------------------------------------------------------------------
#define LROW 40                      // bf16 elems per LDS row (32 data + pad)
#define A_ELEMS (128 * LROW)         // per buffer
#define B_ELEMS (64 * LROW)

__global__ void __launch_bounds__(128)
proj_kernel(const uint16_t* __restrict__ A,
            const uint16_t* __restrict__ Wq,
            const uint16_t* __restrict__ Wk,
            const float* __restrict__ bq,
            const float* __restrict__ bk,
            float* __restrict__ Cq,
            float* __restrict__ Ck)
{
  __shared__ uint16_t smem[2 * A_ELEMS + 2 * B_ELEMS];   // 30 KB

  const int K = EE, N = EE;
  const uint16_t* Wm   = blockIdx.z ? Wk : Wq;
  const float*    bias = blockIdx.z ? bk : bq;
  float*          C    = blockIdx.z ? Ck : Cq;

  int tid   = threadIdx.x;
  int wv    = tid >> 5;
  int lane  = tid & 31;
  int row16 = lane & 15;
  int half  = lane >> 4;
  int nBaseWG = blockIdx.x * 64;
  int mBaseWG = blockIdx.y * 128;
  int nW = nBaseWG + (wv & 1) * 32;       // wave N origin
  int mW = mBaseWG + (wv >> 1) * 64;      // wave M origin

  const v8f vzero = {0.f,0.f,0.f,0.f,0.f,0.f,0.f,0.f};
  v8f acc[4][2];
#pragma unroll
  for (int i = 0; i < 4; ++i)
#pragma unroll
    for (int j = 0; j < 2; ++j) acc[i][j] = vzero;

  // cooperative copy descriptors (per thread: 4 A-chunks + 2 B-chunks of 16B)
  const uint16_t* aSrcRow = A  + (size_t)(mBaseWG + tid) * K;          // row = tid
  const uint16_t* bSrcRow = Wm + (size_t)(nBaseWG + (tid & 63)) * K;
  int bChunk0 = (tid >> 6) * 2;
  uint16_t* aDstRow = smem + tid * LROW;                 // + buf*A_ELEMS
  uint16_t* bDstRow = smem + 2 * A_ELEMS + (tid & 63) * LROW;

  // issue copy of K-slab k0 into buffer p
  auto issue_tile = [&](int k0, int p) {
#pragma unroll
    for (int c = 0; c < 4; ++c)
      copy16_g2l(aDstRow + p * A_ELEMS + c * 8, aSrcRow + k0 + c * 8);
#pragma unroll
    for (int c = 0; c < 2; ++c) {
      int ch = bChunk0 + c;
      copy16_g2l(bDstRow + p * B_ELEMS + ch * 8, bSrcRow + k0 + ch * 8);
    }
  };

  int p = 0;
  issue_tile(0, p);

  for (int k0 = 0; k0 < K; k0 += 32) {
    int more = (k0 + 32 < K);
    if (more) issue_tile(k0 + 32, p ^ 1);
    wait_async_le(more);          // previous buffer's 6 async ops complete
    __syncthreads();              // whole WG sees buffer p filled

    const uint16_t* aB = smem + p * A_ELEMS;
    const uint16_t* bB = smem + 2 * A_ELEMS + p * B_ELEMS;
    Frag a[4], bf_[2];
#pragma unroll
    for (int ms = 0; ms < 4; ++ms) {
      const uint16_t* q = aB + ((mW - mBaseWG) + ms * 16 + row16) * LROW + half * 8;
      a[ms].u[0] = *(const uint4*)q;
      a[ms].u[1] = *(const uint4*)(q + 16);
    }
#pragma unroll
    for (int ns = 0; ns < 2; ++ns) {
      const uint16_t* q = bB + ((nW - nBaseWG) + ns * 16 + row16) * LROW + half * 8;
      bf_[ns].u[0] = *(const uint4*)q;
      bf_[ns].u[1] = *(const uint4*)(q + 16);
    }
#pragma unroll
    for (int ms = 0; ms < 4; ++ms)
#pragma unroll
      for (int ns = 0; ns < 2; ++ns)
        acc[ms][ns] = __builtin_amdgcn_wmma_f32_16x16x32_bf16(
            false, a[ms].v, false, bf_[ns].v, (short)0, acc[ms][ns], false, false);

    __syncthreads();              // buffer p free for the next overwrite
    p ^= 1;
  }

#pragma unroll
  for (int ns = 0; ns < 2; ++ns) {
    float bv = bias[nW + ns*16 + row16];
    int colg = nW + ns*16 + row16;
#pragma unroll
    for (int ms = 0; ms < 4; ++ms)
#pragma unroll
      for (int r = 0; r < 8; ++r) {
        int rowg = mW + ms*16 + r + half*8;
        C[(size_t)rowg * N + colg] = acc[ms][ns][r] + bv;
      }
  }
}

// ---------------------------------------------------------------------------
// Kernel 4: RoPE (rotate-every-two on first 64 dims) + (B,S,H,D)->(B,H,S,D)
// transpose; writes f32 result to d_out and a bf16 shadow for the score WMMA.
// ---------------------------------------------------------------------------
__global__ void __launch_bounds__(256)
rope_kernel(const float* __restrict__ tin,   // (B,S,H,D)
            const float* __restrict__ ep,    // (S, 64) = [sin(32) | cos(32)]
            const int*   __restrict__ pid,   // (B,S)
            float*       __restrict__ tout,  // (B,H,S,D) f32
            uint16_t*    __restrict__ tbf)   // (B,H,S,D) bf16
{
  unsigned tid = blockIdx.x * blockDim.x + threadIdx.x;
  int d2 = tid & 127;
  int h  = (tid >> 7)  & 15;
  int s  = (tid >> 11) & 2047;
  int b  = (tid >> 22) & 1;
  size_t inOff = (((size_t)(b * SS + s) * HH + h) * DD) + d2 * 2;
  float2 v = *(const float2*)(tin + inOff);
  float o0 = v.x, o1 = v.y;
  if (d2 < 32) {
    int pos = pid[b * SS + s];
    float sn = ep[pos * RR + d2];
    float cs = ep[pos * RR + 32 + d2];
    o0 = v.x * cs - v.y * sn;
    o1 = v.y * cs + v.x * sn;
  }
  size_t outOff = (((size_t)(b * HH + h) * SS + s) * DD) + d2 * 2;
  *(float2*)(tout + outOff) = make_float2(o0, o1);
  uint32_t p = (uint32_t)f2bf(o0) | ((uint32_t)f2bf(o1) << 16);
  *(uint32_t*)(tbf + outOff) = p;
}

// ---------------------------------------------------------------------------
// Kernel 5: causal scores + softmax. WG = 2 waves, 16 q-rows per wave,
// full 2048-wide score strip kept in 256 KB LDS (single pass over HBM for
// the 537 MB attn output). bf16 WMMA for q.k^T, fully-masked tiles skipped.
// ---------------------------------------------------------------------------
__global__ void __launch_bounds__(64)
attn_kernel(const uint16_t* __restrict__ qb,  // (B,H,S,D) bf16
            const uint16_t* __restrict__ kb,  // (B,H,S,D) bf16
            float* __restrict__ attn)         // (B,H,S,S) f32
{
  extern __shared__ float smem[];             // 2 * 16 * 2048 floats
  int wv    = threadIdx.x >> 5;
  int lane  = threadIdx.x & 31;
  int row16 = lane & 15;
  int half  = lane >> 4;
  int bh    = blockIdx.y;
  int qrow0 = blockIdx.x * 32 + wv * 16;

  const uint16_t* qbase = qb + ((size_t)bh * SS + qrow0) * DD;
  const uint16_t* kbase = kb + (size_t)bh * SS * DD;

  Frag a[8];
#pragma unroll
  for (int c = 0; c < 8; ++c) {
    const uint16_t* p = qbase + row16 * DD + c * 32 + half * 8;
    a[c].u[0] = *(const uint4*)p;
    a[c].u[1] = *(const uint4*)(p + 16);
  }

  float* srow = smem + wv * 16 * SS;
  const v8f vzero = {0.f,0.f,0.f,0.f,0.f,0.f,0.f,0.f};
  int ntiles = (qrow0 >> 4) + 1;              // causal: skip fully-masked tiles

  for (int t = 0; t < ntiles; ++t) {
    v8f acc = vzero;
    const uint16_t* kt = kbase + (size_t)(t * 16 + row16) * DD;
#pragma unroll
    for (int c = 0; c < 8; ++c) {
      Frag bf_;
      const uint16_t* p = kt + c * 32 + half * 8;
      bf_.u[0] = *(const uint4*)p;
      bf_.u[1] = *(const uint4*)(p + 16);
      acc = __builtin_amdgcn_wmma_f32_16x16x32_bf16(
          false, a[c].v, false, bf_.v, (short)0, acc, false, false);
    }
#pragma unroll
    for (int r = 0; r < 8; ++r) {
      int rowl = r + half * 8;
      int rowg = qrow0 + rowl;
      int colg = t * 16 + row16;
      float v = acc[r] * 0.0625f;             // 1/sqrt(256)
      if (colg > rowg) v = -3.0e38f;          // causal mask on diagonal tile
      srow[rowl * SS + colg] = v;
    }
  }

  int ncols = ntiles * 16;
  float* attb = attn + (size_t)bh * SS * SS;
  for (int r = 0; r < 16; ++r) {
    int rowg = qrow0 + r;
    float* rp = srow + r * SS;
    float m = -3.4e38f;
    for (int c = lane; c < ncols; c += 32) m = fmaxf(m, rp[c]);
#pragma unroll
    for (int d = 16; d >= 1; d >>= 1) m = fmaxf(m, __shfl_xor(m, d, 32));
    float sum = 0.f;
    for (int c = lane; c < ncols; c += 32) {
      float e = __expf(rp[c] - m);
      rp[c] = e;
      sum += e;
    }
#pragma unroll
    for (int d = 16; d >= 1; d >>= 1) sum += __shfl_xor(sum, d, 32);
    float inv = 1.0f / sum;
    float* op = attb + (size_t)rowg * SS;
    for (int c = lane; c < ncols; c += 32) op[c] = rp[c] * inv;
    for (int c = ncols + lane; c < SS; c += 32) op[c] = 0.0f;
  }
}

// ---------------------------------------------------------------------------
extern "C" void kernel_launch(void* const* d_in, const int* in_sizes, int n_in,
                              void* d_out, int out_size, void* d_ws, size_t ws_size,
                              hipStream_t stream) {
  (void)in_sizes; (void)n_in; (void)out_size; (void)ws_size;
  const float* x    = (const float*)d_in[0];
  const float* q_w  = (const float*)d_in[1];
  const float* q_b  = (const float*)d_in[2];
  const float* k_w  = (const float*)d_in[3];
  const float* k_b  = (const float*)d_in[4];
  const float* ln_w = (const float*)d_in[5];
  const float* ln_b = (const float*)d_in[6];
  const float* ep   = (const float*)d_in[7];
  const int*   pid  = (const int*)d_in[8];

  char* ws = (char*)d_ws;
  uint16_t* xn   = (uint16_t*)ws;  ws += (size_t)MTOT * EE * 2;   // x_norm bf16
  uint16_t* qwb  = (uint16_t*)ws;  ws += (size_t)EE * EE * 2;     // q_w bf16
  uint16_t* kwb  = (uint16_t*)ws;  ws += (size_t)EE * EE * 2;     // k_w bf16
  float*    qtmp = (float*)ws;     ws += (size_t)MTOT * EE * 4;   // q (B,S,H,D)
  float*    ktmp = (float*)ws;     ws += (size_t)MTOT * EE * 4;   // k (B,S,H,D)
  uint16_t* qbf  = (uint16_t*)ws;  ws += (size_t)MTOT * EE * 2;   // q bf16 (B,H,S,D)
  uint16_t* kbf  = (uint16_t*)ws;  ws += (size_t)MTOT * EE * 2;   // k bf16 (B,H,S,D)

  float* qout = (float*)d_out;                        // (B,H,S,D)
  float* kout = qout + (size_t)BB * HH * SS * DD;
  float* attn = kout + (size_t)BB * HH * SS * DD;     // (B,H,S,S)

  ln_kernel<<<MTOT / 8, 256, 0, stream>>>(x, ln_w, ln_b, xn);
  cvt_kernel<<<(EE * EE / 4) / 256, 256, 0, stream>>>(q_w, qwb);
  cvt_kernel<<<(EE * EE / 4) / 256, 256, 0, stream>>>(k_w, kwb);
  proj_kernel<<<dim3(EE / 64, MTOT / 128, 2), 128, 0, stream>>>(
      xn, qwb, kwb, q_b, k_b, qtmp, ktmp);
  rope_kernel<<<(BB * SS * HH * 128) / 256, 256, 0, stream>>>(qtmp, ep, pid, qout, qbf);
  rope_kernel<<<(BB * SS * HH * 128) / 256, 256, 0, stream>>>(ktmp, ep, pid, kout, kbf);
  attn_kernel<<<dim3(SS / 32, BB * HH), 64, 2 * 16 * SS * sizeof(float), stream>>>(
      qbf, kbf, attn);
}
